// CausalSelfAttention_62852551409942
// MI455X (gfx1250) — matrix operand order repaired and checked by reading, not compiled, not run on hardware
//
#include <hip/hip_runtime.h>
#include <hip/hip_bf16.h>

// ---------------------------------------------------------------------------
// Causal self-attention, all matmuls on v_wmma_f32_16x16x32_bf16 (CDNA5).
//   B=4, T=2048, D=1024, H=16, HD=64
// Pipeline: GEMM1 (x@Wqkv -> bf16 Q/K/V in [B,H,T,64], Q pre-scaled by 0.125)
//           flash-attention (64q x 64k tiles, online softmax, WMMA QK^T & PV)
//           GEMM2 (y@Wproj -> f32 out)
// ---------------------------------------------------------------------------

#define B_ 4
#define T_ 2048
#define D_ 1024
#define H_ 16
#define HD_ 64

typedef __attribute__((ext_vector_type(8)))  float        v8f;
typedef __attribute__((ext_vector_type(16))) __bf16       v16bf;
typedef __attribute__((ext_vector_type(4)))  float        f32x4;
typedef __attribute__((ext_vector_type(4)))  unsigned int u32x4;

union FragBf {
    u32x4 q[2];   // two 16-byte chunks
    v16bf v;      // 16 bf16 = 32 bytes (8 VGPRs/lane)
};

// hardware f32 -> bf16 convert (backend emits v_cvt, often packed)
__device__ __forceinline__ unsigned short f2bf(float f) {
    union { __bf16 h; unsigned short u; } c;
    c.h = (__bf16)f;
    return c.u;
}

__device__ __forceinline__ v8f wmma_bf16(v16bf a, v16bf b, v8f c) {
    return __builtin_amdgcn_wmma_f32_16x16x32_bf16(
        false, a, false, b, (short)0, c, false, false);
}

// ---------------------------------------------------------------------------
// Tiled GEMM: C[M,N] = A[M,K] @ Bw[K,N] + bias.  BM=128 BN=64 BK=64.
// 256 threads = 8 waves in a 4(m) x 2(n) grid; each wave -> 32x32 output
// (2x2 WMMA accumulators, 8 WMMAs per K-stage).
// EPI==0 : scatter into Q/K/V [B,H,T,64] bf16 (Q scaled by 0.125)
// EPI==1 : f32 store to Out[M,N]
// ---------------------------------------------------------------------------
template <bool AF32, int EPI>
__global__ __launch_bounds__(256) void gemm_bf16_wmma(
    const void* __restrict__ Aptr, const float* __restrict__ Bw,
    const float* __restrict__ bias, int M, int N, int K,
    unsigned short* __restrict__ Qg, unsigned short* __restrict__ Kg,
    unsigned short* __restrict__ Vg, float* __restrict__ Out)
{
    constexpr int BM = 128, BN = 64, BK = 64, LDA = 72, LDB = 72;
    __shared__ unsigned short sA[BM][LDA];   // [m][k]
    __shared__ unsigned short sB[BN][LDB];   // [n][k]  (B^T so frags are contiguous)

    const int tid  = threadIdx.x;
    const int lane = tid & 31;
    const int wave = tid >> 5;
    const int wm   = wave >> 1;   // 0..3
    const int wn   = wave & 1;    // 0..1
    const int m0   = blockIdx.y * BM;
    const int n0   = blockIdx.x * BN;

    // staging maps (used both for loads and prefetch of the next tile)
    const int arow = tid >> 1;              // 0..127
    const int akb  = (tid & 1) * 32;        // 0 / 32
    const int bkk  = tid >> 2;              // 0..63
    const int bnb  = (tid & 3) * 16;        // 0..48

    v8f c[2][2];
    const v8f vz = {0.f, 0.f, 0.f, 0.f, 0.f, 0.f, 0.f, 0.f};
    c[0][0] = vz; c[0][1] = vz; c[1][0] = vz; c[1][1] = vz;

    for (int k0 = 0; k0 < K; k0 += BK) {
        __syncthreads();
        // ---- stage A tile (128x64) -> bf16 LDS
        if constexpr (AF32) {
            const float* Ag = (const float*)Aptr;
            const f32x4* src =
                (const f32x4*)(Ag + (size_t)(m0 + arow) * K + k0 + akb);
#pragma unroll
            for (int i = 0; i < 8; ++i) {
                f32x4 v = src[i];
#pragma unroll
                for (int t = 0; t < 4; ++t)
                    sA[arow][akb + i * 4 + t] = f2bf(v[t]);
            }
        } else {
            const unsigned short* Ag = (const unsigned short*)Aptr;
            const u32x4* src =
                (const u32x4*)(Ag + (size_t)(m0 + arow) * K + k0 + akb);
            u32x4* dst = (u32x4*)&sA[arow][akb];
#pragma unroll
            for (int i = 0; i < 4; ++i) dst[i] = src[i];
        }
        // ---- stage B tile (64x64) transposed -> sB[n][k]
        {
            const f32x4* src =
                (const f32x4*)(Bw + (size_t)(k0 + bkk) * N + n0 + bnb);
#pragma unroll
            for (int i = 0; i < 4; ++i) {
                f32x4 v = src[i];
#pragma unroll
                for (int t = 0; t < 4; ++t)
                    sB[bnb + i * 4 + t][bkk] = f2bf(v[t]);
            }
        }
        // ---- prefetch next K-tile into cache (global_prefetch_b8)
        if (k0 + BK < K) {
            if constexpr (AF32) {
                const float* Ag = (const float*)Aptr;
                __builtin_prefetch(Ag + (size_t)(m0 + arow) * K + k0 + BK + akb,
                                   0, 1);
            } else {
                const unsigned short* Ag = (const unsigned short*)Aptr;
                __builtin_prefetch(Ag + (size_t)(m0 + arow) * K + k0 + BK + akb,
                                   0, 1);
            }
            __builtin_prefetch(Bw + (size_t)(k0 + BK + bkk) * N + n0 + bnb,
                               0, 1);
        }
        __syncthreads();

        // ---- 2 K-steps of 32, 4 WMMA each -> 8 WMMA per wave per stage
#pragma unroll
        for (int f = 0; f < 2; ++f) {
            FragBf a[2], b[2];
            const int kbaseA = f * 32 + ((lane >> 4) << 3);   // +0/+8
            const int kbaseB = f * 32 + ((lane >> 4) << 3);   // +0/+8
#pragma unroll
            for (int i = 0; i < 2; ++i) {
                const int row = wm * 32 + i * 16 + (lane & 15);
                a[i].q[0] = *(const u32x4*)&sA[row][kbaseA];
                a[i].q[1] = *(const u32x4*)&sA[row][kbaseA + 16];
            }
#pragma unroll
            for (int j = 0; j < 2; ++j) {
                const int col = wn * 32 + j * 16 + (lane & 15);
                b[j].q[0] = *(const u32x4*)&sB[col][kbaseB];
                b[j].q[1] = *(const u32x4*)&sB[col][kbaseB + 16];
            }
#pragma unroll
            for (int i = 0; i < 2; ++i)
#pragma unroll
                for (int j = 0; j < 2; ++j)
                    c[i][j] = wmma_bf16(a[i].v, b[j].v, c[i][j]);
        }
    }

    // ---- epilogue: C layout is M=(lane/16)*8+r (VGPR r), N=lane%16
    const int mrow = (lane >> 4) << 3;
#pragma unroll
    for (int i = 0; i < 2; ++i) {
#pragma unroll
        for (int j = 0; j < 2; ++j) {
            const int nn = n0 + wn * 32 + j * 16 + (lane & 15);
            const float bv = bias[nn];
#pragma unroll
            for (int r = 0; r < 8; ++r) {
                const int mm  = m0 + wm * 32 + i * 16 + mrow + r;
                float     val = c[i][j][r] + bv;
                if constexpr (EPI == 0) {
                    const int part = nn >> 10;          // 0=q 1=k 2=v
                    const int h    = (nn & 1023) >> 6;
                    const int hd   = nn & 63;
                    const int bi   = mm >> 11;          // / T
                    const int t    = mm & (T_ - 1);
                    const size_t off =
                        ((size_t)(bi * H_ + h) * T_ + t) * HD_ + hd;
                    unsigned short* dst = (part == 0) ? Qg
                                         : (part == 1) ? Kg : Vg;
                    if (part == 0) val *= 0.125f;       // 1/sqrt(HD), exact
                    dst[off] = f2bf(val);
                } else {
                    Out[(size_t)mm * N + nn] = val;
                }
            }
        }
    }
}

// ---------------------------------------------------------------------------
// Flash attention. 128 threads = 4 waves; each wave owns 16 query rows of a
// 64-row query block. 64-key tiles staged in LDS; online softmax in f32.
// Q is pre-scaled by 0.125, so S = Q@K^T directly.
// ---------------------------------------------------------------------------
__global__ __launch_bounds__(128) void attn_flash_wmma(
    const unsigned short* __restrict__ Qg,
    const unsigned short* __restrict__ Kg,
    const unsigned short* __restrict__ Vg,
    unsigned short* __restrict__ Yw)
{
    __shared__ unsigned short sK [64][72];   // [key][hd]  (= B^T of K^T)
    __shared__ unsigned short sVT[64][72];   // [hd][key]  (= B^T of V)
    __shared__ unsigned short sP [64][72];   // [q_local][key_local]

    const int tid  = threadIdx.x;
    const int lane = tid & 31;
    const int wave = tid >> 5;
    const int qblk = blockIdx.x;      // 0..31
    const int h    = blockIdx.y;
    const int b    = blockIdx.z;
    const int q0   = qblk * 64;
    const size_t headBase = (size_t)(b * H_ + h) * T_ * HD_;

    // Q fragments (16x64 per wave -> two 16x32 A-frags), straight from global
    FragBf qf[2];
    {
        const size_t qrow =
            headBase + (size_t)(q0 + wave * 16 + (lane & 15)) * HD_;
        const int kbase = (lane >> 4) << 3;
#pragma unroll
        for (int f = 0; f < 2; ++f) {
            qf[f].q[0] = *(const u32x4*)(Qg + qrow + f * 32 + kbase);
            qf[f].q[1] = *(const u32x4*)(Qg + qrow + f * 32 + kbase + 16);
        }
    }

    const v8f vz = {0.f, 0.f, 0.f, 0.f, 0.f, 0.f, 0.f, 0.f};
    v8f   o[4] = {vz, vz, vz, vz};
    float rm[8], rl[8];
#pragma unroll
    for (int r = 0; r < 8; ++r) { rm[r] = -1e30f; rl[r] = 0.f; }

    const int Mb = (lane >> 4) << 3;          // C-layout row base per half-wave

    for (int kb = 0; kb <= qblk; ++kb) {
        const int k0g = kb * 64;
        __syncthreads();
        // ---- stage K tile (natural [key][hd])
        {
            const int row = tid >> 1;
            const int cb  = (tid & 1) * 32;
            const u32x4* src =
                (const u32x4*)(Kg + headBase + (size_t)(k0g + row) * HD_ + cb);
            u32x4* dst = (u32x4*)&sK[row][cb];
            dst[0] = src[0]; dst[1] = src[1]; dst[2] = src[2]; dst[3] = src[3];
        }
        // ---- stage V transposed ([hd][key])
        for (int e = tid; e < 4096; e += 128) {
            const int key = e >> 6, hd = e & 63;
            sVT[hd][key] = Vg[headBase + (size_t)(k0g + key) * HD_ + hd];
        }
        // ---- prefetch next key block (global_prefetch_b8)
        if (kb < qblk) {
            const int row = tid >> 1;
            const int cb  = (tid & 1) * 32;
            __builtin_prefetch(
                Kg + headBase + (size_t)(k0g + 64 + row) * HD_ + cb, 0, 1);
            __builtin_prefetch(
                Vg + headBase + (size_t)(k0g + 64 + row) * HD_ + cb, 0, 1);
        }
        __syncthreads();

        // ---- S = Q @ K^T  (4 n-tiles, 2 k-steps)
        v8f s[4] = {vz, vz, vz, vz};
#pragma unroll
        for (int f = 0; f < 2; ++f) {
            const int kbb = f * 32 + ((lane >> 4) << 3);
#pragma unroll
            for (int nt = 0; nt < 4; ++nt) {
                FragBf bf;
                const int col = nt * 16 + (lane & 15);
                bf.q[0] = *(const u32x4*)&sK[col][kbb];
                bf.q[1] = *(const u32x4*)&sK[col][kbb + 16];
                s[nt] = wmma_bf16(qf[f].v, bf.v, s[nt]);
            }
        }

        // ---- causal mask (only the diagonal block)
        if (kb == qblk) {
#pragma unroll
            for (int nt = 0; nt < 4; ++nt) {
                const int kg = k0g + nt * 16 + (lane & 15);
#pragma unroll
                for (int r = 0; r < 8; ++r) {
                    const int qg = q0 + wave * 16 + Mb + r;
                    if (kg > qg) s[nt][r] = -1e30f;
                }
            }
        }

        // ---- online softmax: row stats via 16-lane butterfly
        float mx[8];
#pragma unroll
        for (int r = 0; r < 8; ++r)
            mx[r] = fmaxf(fmaxf(s[0][r], s[1][r]), fmaxf(s[2][r], s[3][r]));
#pragma unroll
        for (int off = 1; off <= 8; off <<= 1)
#pragma unroll
            for (int r = 0; r < 8; ++r)
                mx[r] = fmaxf(mx[r], __shfl_xor(mx[r], off, 32));

        float alpha[8];
#pragma unroll
        for (int r = 0; r < 8; ++r) {
            const float mn = fmaxf(rm[r], mx[r]);
            alpha[r] = __expf(rm[r] - mn);
            rm[r]    = mn;
        }

        float ps[8];
#pragma unroll
        for (int r = 0; r < 8; ++r) ps[r] = 0.f;
#pragma unroll
        for (int nt = 0; nt < 4; ++nt) {
#pragma unroll
            for (int r = 0; r < 8; ++r) {
                const float p = __expf(s[nt][r] - rm[r]);
                ps[r] += p;
                sP[wave * 16 + Mb + r][nt * 16 + (lane & 15)] = f2bf(p);
            }
        }
#pragma unroll
        for (int off = 1; off <= 8; off <<= 1)
#pragma unroll
            for (int r = 0; r < 8; ++r)
                ps[r] += __shfl_xor(ps[r], off, 32);
#pragma unroll
        for (int r = 0; r < 8; ++r) rl[r] = rl[r] * alpha[r] + ps[r];
#pragma unroll
        for (int nt = 0; nt < 4; ++nt)
#pragma unroll
            for (int r = 0; r < 8; ++r) o[nt][r] *= alpha[r];

        // ---- O += P @ V  (wave reads only its own sP rows; DS is in-order)
#pragma unroll
        for (int f = 0; f < 2; ++f) {
            FragBf af;
            const int prow   = wave * 16 + (lane & 15);
            const int kbaseP = f * 32 + ((lane >> 4) << 3);
            af.q[0] = *(const u32x4*)&sP[prow][kbaseP];
            af.q[1] = *(const u32x4*)&sP[prow][kbaseP + 16];
            const int kbb = f * 32 + ((lane >> 4) << 3);
#pragma unroll
            for (int nt = 0; nt < 4; ++nt) {
                FragBf bf;
                const int col = nt * 16 + (lane & 15);
                bf.q[0] = *(const u32x4*)&sVT[col][kbb];
                bf.q[1] = *(const u32x4*)&sVT[col][kbb + 16];
                o[nt] = wmma_bf16(af.v, bf.v, o[nt]);
            }
        }
    }

    // ---- normalize + merge heads: Yw[b*T+t][h*64 + n]
#pragma unroll
    for (int r = 0; r < 8; ++r) {
        const float invl = 1.0f / rl[r];
        const int   t    = q0 + wave * 16 + Mb + r;
#pragma unroll
        for (int nt = 0; nt < 4; ++nt) {
            const size_t off = ((size_t)(b * T_ + t) << 10) +
                               h * HD_ + nt * 16 + (lane & 15);
            Yw[off] = f2bf(o[nt][r] * invl);
        }
    }
}

// ---------------------------------------------------------------------------
extern "C" void kernel_launch(void* const* d_in, const int* in_sizes, int n_in,
                              void* d_out, int out_size, void* d_ws,
                              size_t ws_size, hipStream_t stream)
{
    const float* x      = (const float*)d_in[0];
    const float* W_qkv  = (const float*)d_in[1];
    const float* b_qkv  = (const float*)d_in[2];
    const float* W_proj = (const float*)d_in[3];
    const float* b_proj = (const float*)d_in[4];
    float*       out    = (float*)d_out;

    // workspace carve: Q,K,V,Y bf16 (8M elems = 16 MB each, 64 MB total)
    const size_t qkvN = (size_t)B_ * H_ * T_ * HD_;
    unsigned short* Qg = (unsigned short*)d_ws;
    unsigned short* Kg = Qg + qkvN;
    unsigned short* Vg = Kg + qkvN;
    unsigned short* Yw = Vg + qkvN;

    const int M = B_ * T_;   // 8192

    // 1) qkv = x @ W_qkv + b ; scatter to Q/K/V bf16 (Q pre-scaled)
    gemm_bf16_wmma<true, 0><<<dim3((3 * D_) / 64, M / 128), 256, 0, stream>>>(
        (const void*)x, W_qkv, b_qkv, M, 3 * D_, D_, Qg, Kg, Vg, nullptr);

    // 2) flash attention
    attn_flash_wmma<<<dim3(T_ / 64, H_, B_), 128, 0, stream>>>(Qg, Kg, Vg, Yw);

    // 3) out = y @ W_proj + b
    gemm_bf16_wmma<false, 1><<<dim3(D_ / 64, M / 128), 256, 0, stream>>>(
        (const void*)Yw, W_proj, b_proj, M, D_, D_, nullptr, nullptr, nullptr,
        out);
}
